// DescrptDPA3Next_22402549416333
// MI455X (gfx1250) — matrix operand order repaired and checked by reading, not compiled, not run on hardware
//
#include <hip/hip_runtime.h>
#include <hip/hip_bf16.h>

// ---------------------------------------------------------------------------
// Problem constants (match reference)
// ---------------------------------------------------------------------------
#define NLOC   128
#define NNEI   64
#define ASEL   24
#define NE     (NLOC * NNEI)            // 8192 edges
#define NA     (NLOC * ASEL * ASEL)     // 73728 angles
#define NDIM   128
#define EDIM   128
#define ADIM   64
#define NRAD   12
#define AF     (ADIM + NDIM + 2 * EDIM) // 448
#define EF     (EDIM + 2 * NDIM)        // 384

#define A_SCALE 0.645497224367903f      // (24/10)^-0.5
#define E_SCALE 0.395284707521047f      // (64/10)^-0.5

typedef __attribute__((ext_vector_type(16))) __bf16 v16bf;
typedef __attribute__((ext_vector_type(8)))  __bf16 v8bf;
typedef __attribute__((ext_vector_type(8)))  float  v8f;

__device__ __forceinline__ __bf16 f2bf(float f) {
  union UF { float f; unsigned u; } a;
  a.f = f;
  unsigned r = a.u + 0x7FFFu + ((a.u >> 16) & 1u);   // round-to-nearest-even
  union UB { unsigned short s; __bf16 b; } c;
  c.s = (unsigned short)(r >> 16);
  return c.b;
}

// ---------------------------------------------------------------------------
// WMMA GEMM: C[M,N] f32 = A[M,K] bf16 (row major, lda=K) x B[K,N] bf16 (ldb=N)
// Block: 256 threads = 8 waves arranged 4(M) x 2(N); wave tile 32x64 =
// 2(M) x 4(N) v_wmma_f32_16x16x32_bf16 fragments (8 accumulators).
// Requirements (hold for every GEMM in this pipeline):
//   M multiple of 128 is NOT required (per-wave m-guard), but M multiple of 32, N multiple
//   of 64; K multiple of 32.
// A fragment (16-bit A 16x32): lane L in [0,16) holds row m0+L,
//   elems 0..7 = K kb..kb+7, elems 8..15 = K kb+16..kb+23, kb = k0 (+8 for
//   lanes 16..31).  B fragment (32x16): lane L holds col n0+(L&15), same K
//   order as A.  C/D: lane L -> col n0+(L&15), VGPR v -> row m0+v+((L>>4)<<3).
// ---------------------------------------------------------------------------
__device__ __forceinline__ v16bf load_a_frag(const __bf16* __restrict__ p) {
  v8bf lo = *(const v8bf*)(p);
  v8bf hi = *(const v8bf*)(p + 16);
  v16bf r;
#pragma unroll
  for (int i = 0; i < 8; ++i) { r[i] = lo[i]; r[i + 8] = hi[i]; }
  return r;
}

#define WMMA_BF16(a, b, c) \
  __builtin_amdgcn_wmma_f32_16x16x32_bf16(false, (a), false, (b), (short)0, \
                                          (c), false, false)

__global__ void __launch_bounds__(256) gemm_bf16_wmma_kernel(
    const __bf16* __restrict__ A, const __bf16* __restrict__ B,
    float* __restrict__ C, int M, int N, int K) {
  const int lane = threadIdx.x & 31;
  const int wave = threadIdx.x >> 5;
  const int m0 = blockIdx.y * 128 + (wave >> 1) * 32;
  const int n0 = blockIdx.x * 128 + (wave & 1) * 64;
  if (m0 >= M || n0 >= N) return;   // no barriers in kernel -> safe early exit

  v8f c00 = {}, c01 = {}, c02 = {}, c03 = {};
  v8f c10 = {}, c11 = {}, c12 = {}, c13 = {};

  // Pointer-increment addressing: no per-iteration 64-bit multiplies.
  const int abase = (lane >> 4) << 3;                       // 0 or 8
  const __bf16* ap0 = A + (size_t)(m0 + (lane & 15)) * K + abase;
  const __bf16* ap1 = ap0 + (size_t)16 * K;
  const __bf16* bp  = B + (size_t)abase * N + n0 + (lane & 15);
  const size_t bstep = (size_t)32 * N;

  for (int k = 0; k < K; k += 32) {
    v16bf a0 = load_a_frag(ap0);
    v16bf a1 = load_a_frag(ap1);
    v16bf b0, b1, b2, b3;
#pragma unroll
    for (int i = 0; i < 16; ++i) {
      const __bf16* q = bp + (size_t)(i < 8 ? i : i + 8) * N;
      b0[i] = q[0]; b1[i] = q[16]; b2[i] = q[32]; b3[i] = q[48];
    }
    // Speculative DEV-scope prefetch of next A chunk (dropped if OOB).
    __builtin_prefetch((const void*)(ap0 + 32), 0, 1);
    c00 = WMMA_BF16(a0, b0, c00);
    c01 = WMMA_BF16(a0, b1, c01);
    c02 = WMMA_BF16(a0, b2, c02);
    c03 = WMMA_BF16(a0, b3, c03);
    c10 = WMMA_BF16(a1, b0, c10);
    c11 = WMMA_BF16(a1, b1, c11);
    c12 = WMMA_BF16(a1, b2, c12);
    c13 = WMMA_BF16(a1, b3, c13);
    ap0 += 32;
    ap1 += 32;
    bp += bstep;
  }

  const int ccol = n0 + (lane & 15);
  const int crow = m0 + ((lane >> 4) << 3);
  float* __restrict__ cr0 = C + (size_t)crow * N + ccol;
  float* __restrict__ cr1 = cr0 + (size_t)16 * N;
#pragma unroll
  for (int v = 0; v < 8; ++v) {
    cr0[(size_t)v * N]      = c00[v];
    cr0[(size_t)v * N + 16] = c01[v];
    cr0[(size_t)v * N + 32] = c02[v];
    cr0[(size_t)v * N + 48] = c03[v];
    cr1[(size_t)v * N]      = c10[v];
    cr1[(size_t)v * N + 16] = c11[v];
    cr1[(size_t)v * N + 32] = c12[v];
    cr1[(size_t)v * N + 48] = c13[v];
  }
}

// ---------------------------------------------------------------------------
// Elementwise / reduction helpers
// ---------------------------------------------------------------------------

// RMSNorm rows (f32 in, bf16 out). One wave32 per row.
__global__ void rmsnorm_bf16_kernel(const float* __restrict__ x,
                                    const float* __restrict__ w,
                                    __bf16* __restrict__ y, int R, int D) {
  int wid = (blockIdx.x * blockDim.x + threadIdx.x) >> 5;
  int lane = threadIdx.x & 31;
  if (wid >= R) return;
  const float* xr = x + (size_t)wid * D;
  float s = 0.f;
  for (int d = lane; d < D; d += 32) { float v = xr[d]; s += v * v; }
#pragma unroll
  for (int off = 16; off; off >>= 1) s += __shfl_xor(s, off, 32);
  float inv = rsqrtf(s / (float)D + 1e-6f);
  __bf16* yr = y + (size_t)wid * D;
  for (int d = lane; d < D; d += 32) yr[d] = f2bf(xr[d] * inv * w[d]);
}

// f32 -> bf16 (optionally per-row scale). rowscale==nullptr -> 1.0
__global__ void rows_to_bf16_kernel(__bf16* __restrict__ dst,
                                    const float* __restrict__ src,
                                    const float* __restrict__ rowscale,
                                    long R, int D) {
  long n = R * (long)D;
  for (long i = blockIdx.x * (long)blockDim.x + threadIdx.x; i < n;
       i += (long)gridDim.x * blockDim.x) {
    float s = rowscale ? rowscale[i / D] : 1.f;
    dst[i] = f2bf(src[i] * s);
  }
}

// Build angle feature rows: [a_n(64) | node(128) | e_n[eij](128) | e_n[eik](128)]
__global__ void build_featA_kernel(__bf16* __restrict__ feat,
                                   const __bf16* __restrict__ a_n,
                                   const __bf16* __restrict__ nodeB,
                                   const __bf16* __restrict__ e_n,
                                   const int* __restrict__ n2a,
                                   const int* __restrict__ eij,
                                   const int* __restrict__ eik) {
  int r = blockIdx.x;
  __bf16* f = feat + (size_t)r * AF;
  const __bf16* pa = a_n + (size_t)r * ADIM;
  const __bf16* pn = nodeB + (size_t)n2a[r] * NDIM;
  const __bf16* pi = e_n + (size_t)eij[r] * EDIM;
  const __bf16* pk = e_n + (size_t)eik[r] * EDIM;
  for (int c = threadIdx.x; c < ADIM; c += blockDim.x) f[c] = pa[c];
  for (int c = threadIdx.x; c < NDIM; c += blockDim.x) {
    f[ADIM + c] = pn[c];
    f[ADIM + NDIM + c] = pi[c];
    f[ADIM + NDIM + EDIM + c] = pk[c];
  }
}

// Build edge feature rows: [e_n(128) | n_n[n2e](128) | n_n[next2e](128)]
__global__ void build_featE_kernel(__bf16* __restrict__ feat,
                                   const __bf16* __restrict__ e_n,
                                   const __bf16* __restrict__ n_n,
                                   const int* __restrict__ n2e,
                                   const int* __restrict__ next2e) {
  int r = blockIdx.x;
  __bf16* f = feat + (size_t)r * EF;
  const __bf16* pe = e_n + (size_t)r * EDIM;
  const __bf16* p0 = n_n + (size_t)n2e[r] * NDIM;
  const __bf16* p1 = n_n + (size_t)next2e[r] * NDIM;
  for (int c = threadIdx.x; c < EDIM; c += blockDim.x) f[c] = pe[c];
  for (int c = threadIdx.x; c < NDIM; c += blockDim.x) {
    f[EDIM + c] = p0[c];
    f[EDIM + NDIM + c] = p1[c];
  }
}

// Per-column softmax over contiguous segments of L rows (in-place, pre-scale).
__global__ void seg_softmax_kernel(float* __restrict__ s, int L, int D,
                                   float scale) {
  int seg = blockIdx.x;
  int c = threadIdx.x;               // blockDim == D
  float* p = s + (size_t)seg * L * D + c;
  float m = -3.4e38f;
  for (int i = 0; i < L; ++i) m = fmaxf(m, p[(size_t)i * D] * scale);
  float sum = 0.f;
  for (int i = 0; i < L; ++i) {
    float e = __expf(p[(size_t)i * D] * scale - m);
    p[(size_t)i * D] = e;
    sum += e;
  }
  float inv = 1.f / (sum + 1e-12f);
  for (int i = 0; i < L; ++i) p[(size_t)i * D] *= inv;
}

// GatedMLP epilogue: h[r, 0:H]=v, h[r, H:2H]=g; out = rmsnorm(v*silu(g))*nw.
// blockDim == H == 256.
__global__ void __launch_bounds__(256) gmlp_act_norm_kernel(
    const float* __restrict__ h, const float* __restrict__ nw,
    __bf16* __restrict__ out, int H) {
  int r = blockIdx.x, c = threadIdx.x;
  const float* hr = h + (size_t)r * 2 * H;
  float v = hr[c], g = hr[H + c];
  float x = v * (g / (1.f + __expf(-g)));
  __shared__ float red[8];
  float s = x * x;
#pragma unroll
  for (int off = 16; off; off >>= 1) s += __shfl_xor(s, off, 32);
  if ((threadIdx.x & 31) == 0) red[threadIdx.x >> 5] = s;
  __syncthreads();
  float tot = 0.f;
#pragma unroll
  for (int i = 0; i < 8; ++i) tot += red[i];
  float inv = rsqrtf(tot / (float)H + 1e-6f);
  out[(size_t)r * H + c] = f2bf(x * inv * nw[c]);
}

// Segment reduce over L contiguous rows: dst[idx[seg*L]] (+)= scale*sum(src*mult)
__global__ void seg_reduce_kernel(float* __restrict__ dst,
                                  const float* __restrict__ src,
                                  const float* __restrict__ mult,
                                  const int* __restrict__ idx, int L, int D,
                                  float scale, int accumulate) {
  int seg = blockIdx.x, c = threadIdx.x;  // blockDim == D
  int target = idx[(size_t)seg * L];
  const float* sp = src + (size_t)seg * L * D + c;
  float acc = 0.f;
  if (mult) {
    const float* mp = mult + (size_t)seg * L * D + c;
    for (int i = 0; i < L; ++i) acc += sp[(size_t)i * D] * mp[(size_t)i * D];
  } else {
    for (int i = 0; i < L; ++i) acc += sp[(size_t)i * D];
  }
  acc *= scale;
  float* dp = dst + (size_t)target * D + c;
  if (accumulate) *dp += acc; else *dp = acc;
}

// env = rbf[R,Kr] @ W[Kr,D]  (Kr=12, stays VALU — too small for WMMA)
__global__ void rbf_env_kernel(float* __restrict__ env,
                               const float* __restrict__ rbf,
                               const float* __restrict__ W, int R, int D,
                               int Kr) {
  long n = (long)R * D;
  for (long i = blockIdx.x * (long)blockDim.x + threadIdx.x; i < n;
       i += (long)gridDim.x * blockDim.x) {
    int r = (int)(i / D), c = (int)(i % D);
    const float* rp = rbf + (size_t)r * Kr;
    float acc = 0.f;
    for (int k = 0; k < Kr; ++k) acc += rp[k] * W[(size_t)k * D + c];
    env[i] = acc;
  }
}

// mult[r,c] = env[idx? idx[r] : r, c] * rowscale[r]
__global__ void gate_mult_kernel(float* __restrict__ dst,
                                 const float* __restrict__ env,
                                 const float* __restrict__ rowscale,
                                 const int* __restrict__ idx, long R, int D) {
  long n = R * (long)D;
  for (long i = blockIdx.x * (long)blockDim.x + threadIdx.x; i < n;
       i += (long)gridDim.x * blockDim.x) {
    long r = i / D;
    int c = (int)(i % D);
    long e = idx ? (long)idx[r] : r;
    dst[i] = env[(size_t)e * D + c] * rowscale[r];
  }
}

__global__ void add_inplace_kernel(float* __restrict__ dst,
                                   const float* __restrict__ src, long n) {
  for (long i = blockIdx.x * (long)blockDim.x + threadIdx.x; i < n;
       i += (long)gridDim.x * blockDim.x)
    dst[i] += src[i];
}

__global__ void zero_kernel(float* __restrict__ dst, long n) {
  for (long i = blockIdx.x * (long)blockDim.x + threadIdx.x; i < n;
       i += (long)gridDim.x * blockDim.x)
    dst[i] = 0.f;
}

// ---------------------------------------------------------------------------
// Host orchestration
// ---------------------------------------------------------------------------
static inline dim3 grid1(long n, int b) { return dim3((unsigned)((n + b - 1) / b)); }

extern "C" void kernel_launch(void* const* d_in, const int* in_sizes, int n_in,
                              void* d_out, int out_size, void* d_ws,
                              size_t ws_size, hipStream_t stream) {
  (void)in_sizes; (void)n_in; (void)out_size; (void)ws_size;

  // ---- inputs ----
  const float* in_node  = (const float*)d_in[0];
  const float* in_edge  = (const float*)d_in[1];
  const float* in_angle = (const float*)d_in[2];
  const float* sw       = (const float*)d_in[3];
  const float* a_sw     = (const float*)d_in[4];
  const float* rbf      = (const float*)d_in[5];
  const int*   n2e      = (const int*)d_in[6];
  const int*   next2e   = n2e + NE;
  const int*   n2a      = (const int*)d_in[7];
  const int*   eij2a    = n2a + NA;
  const int*   eik2a    = n2a + 2 * NA;
  const float* la_norm_e = (const float*)d_in[8];
  const float* la_norm_a = (const float*)d_in[9];
  const float* la_win    = (const float*)d_in[10];
  const float* la_nw     = (const float*)d_in[11];
  const float* la_wout   = (const float*)d_in[12];
  const float* la_gate   = (const float*)d_in[13];
  const float* aa_norm_n = (const float*)d_in[14];
  const float* aa_norm_e = (const float*)d_in[15];
  const float* aa_win    = (const float*)d_in[16];
  const float* aa_nw     = (const float*)d_in[17];
  const float* aa_wout   = (const float*)d_in[18];
  const float* aa_src_gate = (const float*)d_in[19];
  const float* ae_win    = (const float*)d_in[20];
  const float* ae_nw     = (const float*)d_in[21];
  const float* ae_wout   = (const float*)d_in[22];
  const float* lr_norm_e = (const float*)d_in[23];
  const float* lr_norm_a = (const float*)d_in[24];
  const float* lr_win    = (const float*)d_in[25];
  const float* lr_nw     = (const float*)d_in[26];
  const float* lr_wout   = (const float*)d_in[27];
  const float* lr_env    = (const float*)d_in[28];
  const float* lr_eproj  = (const float*)d_in[29];
  const float* lr_aproj  = (const float*)d_in[30];
  const float* ar_norm_n = (const float*)d_in[31];
  const float* ar_norm_e = (const float*)d_in[32];
  const float* ar_win    = (const float*)d_in[33];
  const float* ar_nw     = (const float*)d_in[34];
  const float* ar_wout   = (const float*)d_in[35];
  const float* ar_env    = (const float*)d_in[36];
  const float* ar_nproj  = (const float*)d_in[37];
  const float* ar_eproj  = (const float*)d_in[38];

  // ---- live output state ----
  float* out_node  = (float*)d_out;
  float* out_edge  = out_node + (size_t)NLOC * NDIM;
  float* out_angle = out_edge + (size_t)NE * EDIM;
  hipMemcpyAsync(out_node, in_node, (size_t)NLOC * NDIM * 4,
                 hipMemcpyDeviceToDevice, stream);
  hipMemcpyAsync(out_edge, in_edge, (size_t)NE * EDIM * 4,
                 hipMemcpyDeviceToDevice, stream);
  hipMemcpyAsync(out_angle, in_angle, (size_t)NA * ADIM * 4,
                 hipMemcpyDeviceToDevice, stream);

  // ---- workspace carving ----
  char* wsp = (char*)d_ws;
  auto carve = [&](size_t bytes) -> void* {
    void* p = (void*)wsp;
    wsp += (bytes + 255) & ~(size_t)255;
    return p;
  };
  __bf16* featA   = (__bf16*)carve((size_t)NA * AF * 2);
  __bf16* featE   = (__bf16*)carve((size_t)NE * EF * 2);
  float*  hiddenF = (float*)carve((size_t)NA * 512 * 4);   // also Ctmp
  __bf16* hiddenN = (__bf16*)carve((size_t)NA * 256 * 2);
  float*  scoresF = (float*)carve((size_t)NA * 128 * 4);   // scores / gate-mult
  float*  msgF    = (float*)carve((size_t)NA * 128 * 4);
  __bf16* msgB    = (__bf16*)carve((size_t)NA * 128 * 2);
  __bf16* a_nB    = (__bf16*)carve((size_t)NA * ADIM * 2);
  __bf16* e_nB    = (__bf16*)carve((size_t)NE * EDIM * 2);
  __bf16* n_nB    = (__bf16*)carve((size_t)NLOC * NDIM * 2);
  __bf16* nodeB   = (__bf16*)carve((size_t)NLOC * NDIM * 2);
  float*  e_segF  = (float*)carve((size_t)NE * EDIM * 4);
  float*  n_segF  = (float*)carve((size_t)NLOC * NDIM * 4);
  __bf16* segB    = (__bf16*)carve((size_t)NE * EDIM * 2);
  float*  envF    = (float*)carve((size_t)NE * 128 * 4);
  __bf16* W_la_win  = (__bf16*)carve((size_t)AF * 512 * 2);
  __bf16* W_la_wout = (__bf16*)carve((size_t)256 * 128 * 2);
  __bf16* W_la_gate = (__bf16*)carve((size_t)AF * 128 * 2);
  __bf16* W_aa_win  = (__bf16*)carve((size_t)EF * 512 * 2);
  __bf16* W_aa_wout = (__bf16*)carve((size_t)256 * 128 * 2);
  __bf16* W_aa_gate = (__bf16*)carve((size_t)128 * 128 * 2);
  __bf16* W_ae_win  = (__bf16*)carve((size_t)EF * 512 * 2);
  __bf16* W_ae_wout = (__bf16*)carve((size_t)256 * 128 * 2);
  __bf16* W_lr_win  = (__bf16*)carve((size_t)AF * 512 * 2);
  __bf16* W_lr_wout = (__bf16*)carve((size_t)256 * 128 * 2);
  __bf16* W_lr_ep   = (__bf16*)carve((size_t)128 * 128 * 2);
  __bf16* W_lr_ap   = (__bf16*)carve((size_t)128 * 64 * 2);
  __bf16* W_ar_win  = (__bf16*)carve((size_t)EF * 512 * 2);
  __bf16* W_ar_wout = (__bf16*)carve((size_t)256 * 128 * 2);
  __bf16* W_ar_np   = (__bf16*)carve((size_t)128 * 128 * 2);
  __bf16* W_ar_ep   = (__bf16*)carve((size_t)128 * 128 * 2);
  float*  Ctmp = hiddenF;  // reuse big f32 buffer for GEMM projection outputs

  auto toBf = [&](__bf16* dst, const float* src, long n) {
    rows_to_bf16_kernel<<<grid1(n, 256), 256, 0, stream>>>(dst, src, nullptr,
                                                           n, 1);
  };
  // weight conversions
  toBf(W_la_win, la_win, (long)AF * 512);
  toBf(W_la_wout, la_wout, 256 * 128);
  toBf(W_la_gate, la_gate, (long)AF * 128);
  toBf(W_aa_win, aa_win, (long)EF * 512);
  toBf(W_aa_wout, aa_wout, 256 * 128);
  toBf(W_aa_gate, aa_src_gate, 128 * 128);
  toBf(W_ae_win, ae_win, (long)EF * 512);
  toBf(W_ae_wout, ae_wout, 256 * 128);
  toBf(W_lr_win, lr_win, (long)AF * 512);
  toBf(W_lr_wout, lr_wout, 256 * 128);
  toBf(W_lr_ep, lr_eproj, 128 * 128);
  toBf(W_lr_ap, lr_aproj, 128 * 64);
  toBf(W_ar_win, ar_win, (long)EF * 512);
  toBf(W_ar_wout, ar_wout, 256 * 128);
  toBf(W_ar_np, ar_nproj, 128 * 128);
  toBf(W_ar_ep, ar_eproj, 128 * 128);

  auto gemm = [&](const __bf16* A, const __bf16* B, float* C, int M, int N,
                  int K) {
    dim3 g((N + 127) / 128, (M + 127) / 128);
    gemm_bf16_wmma_kernel<<<g, 256, 0, stream>>>(A, B, C, M, N, K);
  };
  auto rms = [&](const float* x, const float* w, __bf16* y, int R, int D) {
    rmsnorm_bf16_kernel<<<grid1((long)R * 32, 256), 256, 0, stream>>>(x, w, y,
                                                                      R, D);
  };

  // ======== sub-block 1: line-graph attention (angle -> edge) ========
  rms(out_edge, la_norm_e, e_nB, NE, EDIM);
  rms(out_angle, la_norm_a, a_nB, NA, ADIM);
  rows_to_bf16_kernel<<<grid1((long)NLOC * NDIM, 256), 256, 0, stream>>>(
      nodeB, out_node, nullptr, NLOC, NDIM);
  build_featA_kernel<<<NA, 128, 0, stream>>>(featA, a_nB, nodeB, e_nB, n2a,
                                             eij2a, eik2a);
  gemm(featA, W_la_gate, scoresF, NA, 128, AF);
  seg_softmax_kernel<<<NA / ASEL, 128, 0, stream>>>(scoresF, ASEL, 128,
                                                    A_SCALE);
  gemm(featA, W_la_win, hiddenF, NA, 512, AF);
  gmlp_act_norm_kernel<<<NA, 256, 0, stream>>>(hiddenF, la_nw, hiddenN, 256);
  gemm(hiddenN, W_la_wout, msgF, NA, 128, 256);
  seg_reduce_kernel<<<NA / ASEL, 128, 0, stream>>>(out_edge, msgF, scoresF,
                                                   eij2a, ASEL, 128, 1.f, 1);

  // ======== sub-block 2: atom-graph attention (edge -> node, + edge) ========
  rms(out_node, aa_norm_n, n_nB, NLOC, NDIM);
  rms(out_edge, aa_norm_e, e_nB, NE, EDIM);
  build_featE_kernel<<<NE, 128, 0, stream>>>(featE, e_nB, n_nB, n2e, next2e);
  gemm(e_nB, W_aa_gate, scoresF, NE, 128, 128);
  seg_softmax_kernel<<<NLOC, 128, 0, stream>>>(scoresF, NNEI, 128, E_SCALE);
  gemm(featE, W_aa_win, hiddenF, NE, 512, EF);
  gmlp_act_norm_kernel<<<NE, 256, 0, stream>>>(hiddenF, aa_nw, hiddenN, 256);
  gemm(hiddenN, W_aa_wout, msgF, NE, 128, 256);
  seg_reduce_kernel<<<NLOC, 128, 0, stream>>>(out_node, msgF, scoresF, n2e,
                                              NNEI, 128, 1.f, 1);
  gemm(featE, W_ae_win, hiddenF, NE, 512, EF);
  gmlp_act_norm_kernel<<<NE, 256, 0, stream>>>(hiddenF, ae_nw, hiddenN, 256);
  gemm(hiddenN, W_ae_wout, msgF, NE, 128, 256);
  add_inplace_kernel<<<grid1((long)NE * 128, 256), 256, 0, stream>>>(
      out_edge, msgF, (long)NE * 128);

  // ======== sub-block 3: line-graph refinement ========
  rms(out_edge, lr_norm_e, e_nB, NE, EDIM);
  rms(out_angle, lr_norm_a, a_nB, NA, ADIM);
  rows_to_bf16_kernel<<<grid1((long)NLOC * NDIM, 256), 256, 0, stream>>>(
      nodeB, out_node, nullptr, NLOC, NDIM);
  build_featA_kernel<<<NA, 128, 0, stream>>>(featA, a_nB, nodeB, e_nB, n2a,
                                             eij2a, eik2a);
  gemm(featA, W_lr_win, hiddenF, NA, 512, AF);
  gmlp_act_norm_kernel<<<NA, 256, 0, stream>>>(hiddenF, lr_nw, hiddenN, 256);
  gemm(hiddenN, W_lr_wout, msgF, NA, 128, 256);
  rbf_env_kernel<<<grid1((long)NE * 128, 256), 256, 0, stream>>>(
      envF, rbf, lr_env, NE, 128, NRAD);
  gate_mult_kernel<<<grid1((long)NA * 128, 256), 256, 0, stream>>>(
      scoresF, envF, a_sw, eik2a, NA, 128);
  zero_kernel<<<grid1((long)NE * 128, 256), 256, 0, stream>>>(e_segF,
                                                              (long)NE * 128);
  seg_reduce_kernel<<<NA / ASEL, 128, 0, stream>>>(e_segF, msgF, scoresF,
                                                   eij2a, ASEL, 128, A_SCALE,
                                                   0);
  rows_to_bf16_kernel<<<grid1((long)NA * 128, 256), 256, 0, stream>>>(
      msgB, msgF, a_sw, NA, 128);
  rows_to_bf16_kernel<<<grid1((long)NE * 128, 256), 256, 0, stream>>>(
      segB, e_segF, nullptr, NE, 128);
  gemm(segB, W_lr_ep, Ctmp, NE, 128, 128);
  add_inplace_kernel<<<grid1((long)NE * 128, 256), 256, 0, stream>>>(
      out_edge, Ctmp, (long)NE * 128);
  gemm(msgB, W_lr_ap, Ctmp, NA, 64, 128);
  add_inplace_kernel<<<grid1((long)NA * 64, 256), 256, 0, stream>>>(
      out_angle, Ctmp, (long)NA * 64);

  // ======== sub-block 4: atom-graph refinement ========
  rms(out_node, ar_norm_n, n_nB, NLOC, NDIM);
  rms(out_edge, ar_norm_e, e_nB, NE, EDIM);
  build_featE_kernel<<<NE, 128, 0, stream>>>(featE, e_nB, n_nB, n2e, next2e);
  gemm(featE, W_ar_win, hiddenF, NE, 512, EF);
  gmlp_act_norm_kernel<<<NE, 256, 0, stream>>>(hiddenF, ar_nw, hiddenN, 256);
  gemm(hiddenN, W_ar_wout, msgF, NE, 128, 256);
  rbf_env_kernel<<<grid1((long)NE * 128, 256), 256, 0, stream>>>(
      envF, rbf, ar_env, NE, 128, NRAD);
  gate_mult_kernel<<<grid1((long)NE * 128, 256), 256, 0, stream>>>(
      scoresF, envF, sw, nullptr, NE, 128);
  seg_reduce_kernel<<<NLOC, 128, 0, stream>>>(n_segF, msgF, scoresF, n2e,
                                              NNEI, 128, E_SCALE, 0);
  rows_to_bf16_kernel<<<grid1((long)NLOC * 128, 256), 256, 0, stream>>>(
      segB, n_segF, nullptr, NLOC, 128);
  gemm(segB, W_ar_np, Ctmp, NLOC, 128, 128);
  add_inplace_kernel<<<grid1((long)NLOC * 128, 256), 256, 0, stream>>>(
      out_node, Ctmp, (long)NLOC * 128);
  rows_to_bf16_kernel<<<grid1((long)NE * 128, 256), 256, 0, stream>>>(
      msgB, msgF, nullptr, NE, 128);
  gemm(msgB, W_ar_ep, Ctmp, NE, 128, 128);
  add_inplace_kernel<<<grid1((long)NE * 128, 256), 256, 0, stream>>>(
      out_edge, Ctmp, (long)NE * 128);
}
